// InvertedAttention_19894288515403
// MI455X (gfx1250) — matrix operand order repaired and checked by reading, not compiled, not run on hardware
//
#include <hip/hip_runtime.h>
#include <hip/hip_bf16.h>

// ---------------------------------------------------------------------------
// CDNA5 (gfx1250) fused inverted-attention transformer block.
// All GEMMs use v_wmma_f32_16x16x32_bf16 (wave32 WMMA, f32 accumulate).
// ---------------------------------------------------------------------------

typedef __attribute__((ext_vector_type(16))) __bf16 v16bf;
typedef __attribute__((ext_vector_type(8)))  float  v8f;
typedef __attribute__((ext_vector_type(4)))  unsigned u32x4;

union FragA { v16bf v; u32x4 q[2]; };

__device__ inline unsigned short f2bf(float f) {
    union { float f; unsigned u; } v; v.f = f;
    unsigned u = v.u;
    u += 0x7fffu + ((u >> 16) & 1u);   // round-to-nearest-even
    return (unsigned short)(u >> 16);
}
__device__ inline float bf2f(unsigned short h) {
    union { unsigned u; float f; } v; v.u = ((unsigned)h) << 16;
    return v.f;
}
__device__ inline unsigned packbf2(float lo, float hi) {
    return (unsigned)f2bf(lo) | ((unsigned)f2bf(hi) << 16);
}

__device__ inline v8f wmma_bf16(const v16bf& a, const v16bf& b, const v8f& c) {
    return __builtin_amdgcn_wmma_f32_16x16x32_bf16(
        /*neg_a=*/false, a, /*neg_b=*/false, b,
        /*c_mod=*/(short)0, c, /*reuse_a=*/false, /*reuse_b=*/false);
}

// P-buffer index: score (row m, key) -> halfword position inside the
// A-fragment layout of a 16x32 bf16 tile group g = key/32.
__device__ inline int pidx(int m, int key) {
    int g  = key >> 5;
    int kk = key & 31;
    int lp = m + 16 * ((kk >> 3) & 1);
    int ip = (kk & 7) + 8 * (kk >> 4);
    return g * 512 + lp * 16 + ip;
}

// ---------------------------------------------------------------------------
// Kernel 1: QKV projections.  out[r,c] = sum_k xf[r,k]*W[c,k] + b[c]
// Block: 128 threads (4 waves), 64x64 output tile, K swept in steps of 32.
// z = blockIdx.z selects Q / K / V.  Q,K stored bf16 (bt,h,n,dk); V stored
// transposed bf16 (bt,h,dk,n) so the attention PV B-fragments are contiguous.
// ---------------------------------------------------------------------------
__global__ void __launch_bounds__(128)
qkv_proj_kernel(const float* __restrict__ x,
                const float* __restrict__ Wq, const float* __restrict__ bq,
                const float* __restrict__ Wk, const float* __restrict__ bk,
                const float* __restrict__ Wv, const float* __restrict__ bv,
                unsigned short* __restrict__ Qb,
                unsigned short* __restrict__ Kb,
                unsigned short* __restrict__ Vt)
{
    __shared__ __align__(32) unsigned ldsA[1024];   // 4 subtiles * 256 words (frag layout)
    __shared__ __align__(32) unsigned ldsB[1024];

    const int t    = threadIdx.x;
    const int lane = t & 31;
    const int w    = t >> 5;
    const int half = lane >> 4;
    const int n16  = lane & 15;
    const int mbase = blockIdx.x * 64;
    const int nbase = blockIdx.y * 64;
    const int z     = blockIdx.z;
    const float* W    = (z == 0) ? Wq : (z == 1) ? Wk : Wv;
    const float* bias = (z == 0) ? bq : (z == 1) ? bk : bv;

    const v8f vzero = {0.f,0.f,0.f,0.f,0.f,0.f,0.f,0.f};
    v8f acc[4];
    #pragma unroll
    for (int i = 0; i < 4; ++i) acc[i] = vzero;

    for (int ks = 0; ks < 16; ++ks) {
        const int k0 = ks * 32;
        // ---- stage A (x) tile 64x32 f32 -> bf16 fragment layout ----
        #pragma unroll
        for (int c4 = 0; c4 < 4; ++c4) {
            int id = c4 * 128 + t;                  // 0..511 chunk id
            int rr = id >> 3, kc = (id & 7) * 4;
            float4 f = *(const float4*)(x + (size_t)(mbase + rr) * 512 + k0 + kc);
            int s = rr >> 4, r = rr & 15;
            {
                int k = kc;
                int lp = r + 16 * ((k >> 3) & 1);
                ldsA[s * 256 + lp * 8 + ((k & 7) >> 1) + 4 * (k >> 4)] = packbf2(f.x, f.y);
            }
            {
                int k = kc + 2;
                int lp = r + 16 * ((k >> 3) & 1);
                ldsA[s * 256 + lp * 8 + ((k & 7) >> 1) + 4 * (k >> 4)] = packbf2(f.z, f.w);
            }
        }
        // ---- stage B (W) tile 64x32 f32 -> bf16 fragment layout ----
        #pragma unroll
        for (int c4 = 0; c4 < 4; ++c4) {
            int id = c4 * 128 + t;
            int cc = id >> 3, kc = (id & 7) * 4;
            float4 f = *(const float4*)(W + (size_t)(nbase + cc) * 512 + k0 + kc);
            int ct = cc >> 4, n = cc & 15;
            {
                int k = kc;
                int lp = n + 16 * ((k >> 3) & 1);
                ldsB[ct * 256 + lp * 8 + ((k & 7) >> 1) + 4 * (k >> 4)] = packbf2(f.x, f.y);
            }
            {
                int k = kc + 2;
                int lp = n + 16 * ((k >> 3) & 1);
                ldsB[ct * 256 + lp * 8 + ((k & 7) >> 1) + 4 * (k >> 4)] = packbf2(f.z, f.w);
            }
        }
        __syncthreads();
        // ---- compute: wave w owns rows [w*16, w*16+16), all 4 col tiles ----
        const v16bf a = *(const v16bf*)((const unsigned short*)ldsA + w * 512 + lane * 16);
        #pragma unroll
        for (int ct = 0; ct < 4; ++ct) {
            const v16bf b = *(const v16bf*)((const unsigned short*)ldsB + ct * 512 + lane * 16);
            acc[ct] = wmma_bf16(a, b, acc[ct]);
        }
        __syncthreads();
    }

    // ---- epilogue: add bias, store bf16 into attention layouts ----
    #pragma unroll
    for (int ct = 0; ct < 4; ++ct) {
        const int c   = nbase + ct * 16 + n16;     // output feature column
        const float bv_ = bias[c];
        const int h   = c >> 6;
        const int dkk = c & 63;
        const int rbase = mbase + w * 16 + half * 8;   // 8 consecutive rows per lane
        const int bt    = rbase >> 9;                  // 16-row tile never straddles bt
        const int nseq0 = rbase & 511;
        if (z < 2) {
            unsigned short* out = (z == 0) ? Qb : Kb;
            size_t base = ((size_t)bt * 8 + h) * 512 * 64 + (size_t)dkk;
            #pragma unroll
            for (int j = 0; j < 8; ++j)
                out[base + (size_t)(nseq0 + j) * 64] = f2bf(acc[ct][j] + bv_);
        } else {
            u32x4 pk;
            #pragma unroll
            for (int p = 0; p < 4; ++p)
                pk[p] = packbf2(acc[ct][2 * p] + bv_, acc[ct][2 * p + 1] + bv_);
            size_t base = (((size_t)bt * 8 + h) * 64 + dkk) * 512 + nseq0;
            *(u32x4*)(Vt + base) = pk;   // 8 consecutive nseq, 16B store
        }
    }
}

// ---------------------------------------------------------------------------
// Kernel 2: attention for one (bt,h) pair and a 64-query tile.
// scores = Q K^T / 8 -> softmax -> P @ V.  P kept in LDS (bf16, already in
// WMMA A-fragment layout).  Q/K/Vt fragments load directly from global.
// ---------------------------------------------------------------------------
__global__ void __launch_bounds__(128)
attn_kernel(const unsigned short* __restrict__ Qb,
            const unsigned short* __restrict__ Kb,
            const unsigned short* __restrict__ Vt,
            unsigned short* __restrict__ Ctx)
{
    __shared__ __align__(32) unsigned short P[4][8192];   // 16 rows x 512 keys per wave (64KB)

    const int t    = threadIdx.x;
    const int lane = t & 31;
    const int w    = t >> 5;
    const int half = lane >> 4;
    const int n16  = lane & 15;
    const int bh = blockIdx.x;           // 0..1023 = bt*8+h
    const int qt = blockIdx.y;           // 0..7 query tile of 64
    const int bt = bh >> 3, h = bh & 7;
    const size_t head = (size_t)bh * 512 * 64;   // same element count for Q/K and Vt

    const v8f vzero = {0.f,0.f,0.f,0.f,0.f,0.f,0.f,0.f};

    // A-fragments of Q for this wave's 16 rows, dk split 0..31 / 32..63
    const int qrow = qt * 64 + w * 16 + n16;
    const unsigned short* qptr = Qb + head + (size_t)qrow * 64;
    FragA A0, A1;
    A0.q[0] = *(const u32x4*)(qptr + half * 8);
    A0.q[1] = *(const u32x4*)(qptr + 16 + half * 8);
    A1.q[0] = *(const u32x4*)(qptr + 32 + half * 8);
    A1.q[1] = *(const u32x4*)(qptr + 48 + half * 8);

    // ---- phase 1: scores -> scaled bf16 into P (A-frag layout) ----
    for (int kt = 0; kt < 32; ++kt) {
        const int krow = kt * 16 + n16;
        const unsigned short* kptr = Kb + head + (size_t)krow * 64;
        FragA B0, B1;
        B0.q[0] = *(const u32x4*)(kptr + half * 8);
        B0.q[1] = *(const u32x4*)(kptr + 16 + half * 8);
        B1.q[0] = *(const u32x4*)(kptr + 32 + half * 8);
        B1.q[1] = *(const u32x4*)(kptr + 48 + half * 8);
        v8f acc = vzero;
        acc = wmma_bf16(A0.v, B0.v, acc);
        acc = wmma_bf16(A1.v, B1.v, acc);
        #pragma unroll
        for (int j = 0; j < 8; ++j) {
            int m  = j + half * 8;
            int key = kt * 16 + n16;
            P[w][pidx(m, key)] = f2bf(acc[j] * 0.125f);
        }
    }
    __syncthreads();

    // ---- phase 2: softmax in place (lane handles row n16, half of keys) ----
    {
        const int m = n16;
        const int kbase = half * 256;
        float mx = -3.0e38f;
        #pragma unroll 4
        for (int kk = 0; kk < 256; ++kk)
            mx = fmaxf(mx, bf2f(P[w][pidx(m, kbase + kk)]));
        mx = fmaxf(mx, __shfl_xor(mx, 16, 32));
        float sum = 0.f;
        #pragma unroll 4
        for (int kk = 0; kk < 256; ++kk) {
            int idx = pidx(m, kbase + kk);
            float e = __expf(bf2f(P[w][idx]) - mx);
            sum += e;
            P[w][idx] = f2bf(e);
        }
        sum += __shfl_xor(sum, 16, 32);
        const float rs = 1.0f / sum;
        #pragma unroll 4
        for (int kk = 0; kk < 256; ++kk) {
            int idx = pidx(m, kbase + kk);
            P[w][idx] = f2bf(bf2f(P[w][idx]) * rs);
        }
    }
    __syncthreads();

    // ---- phase 3: O = P @ V  (16 x 64 per wave) ----
    v8f o[4];
    #pragma unroll
    for (int i = 0; i < 4; ++i) o[i] = vzero;
    for (int ksx = 0; ksx < 16; ++ksx) {
        const v16bf pa = *(const v16bf*)&P[w][ksx * 512 + lane * 16];
        #pragma unroll
        for (int nt = 0; nt < 4; ++nt) {
            const int vrow = nt * 16 + n16;             // dk row of Vt
            const unsigned short* vp = Vt + head + (size_t)vrow * 512 + ksx * 32 + half * 8;
            FragA B;
            B.q[0] = *(const u32x4*)vp;
            B.q[1] = *(const u32x4*)(vp + 16);
            o[nt] = wmma_bf16(pa, B.v, o[nt]);
        }
    }

    // ---- store context bf16 row-major (65536 x 512) ----
    const int r0 = bt * 512 + qt * 64 + w * 16;
    #pragma unroll
    for (int nt = 0; nt < 4; ++nt) {
        const int c = h * 64 + nt * 16 + n16;
        #pragma unroll
        for (int j = 0; j < 8; ++j) {
            int m = j + half * 8;
            Ctx[(size_t)(r0 + m) * 512 + c] = f2bf(o[nt][j]);
        }
    }
}

// ---------------------------------------------------------------------------
// Kernel 3: out = LN(x + Ctx @ Wo^T + bo).  Block handles 16 full rows so
// LayerNorm statistics stay on-chip.  Wo staged to LDS fragments per k-step;
// the same LDS is reused for the f32 row buffer in the epilogue.
// ---------------------------------------------------------------------------
__global__ void __launch_bounds__(128)
oproj_ln_kernel(const unsigned short* __restrict__ Ctx,
                const float* __restrict__ Wo, const float* __restrict__ bo,
                const float* __restrict__ x,
                const float* __restrict__ gamma, const float* __restrict__ beta,
                float* __restrict__ out)
{
    __shared__ __align__(32) unsigned char smem[16 * 512 * 4];   // 32KB, dual use
    __shared__ float redS[128], redS2[128], stats[32];

    unsigned* ldsB = (unsigned*)smem;          // GEMM phase: Wo fragments
    float*    ldsY = (float*)smem;             // epilogue: 16x512 f32 rows

    const int t    = threadIdx.x;
    const int lane = t & 31;
    const int w    = t >> 5;
    const int half = lane >> 4;
    const int n16  = lane & 15;
    const int rbase = blockIdx.x * 16;

    const v8f vzero = {0.f,0.f,0.f,0.f,0.f,0.f,0.f,0.f};
    v8f acc[8];
    #pragma unroll
    for (int i = 0; i < 8; ++i) acc[i] = vzero;

    const int arow = rbase + n16;
    for (int ks = 0; ks < 16; ++ks) {
        const int k0 = ks * 32;
        // stage Wo tile: all 512 cols x 32 ks -> fragment layout (32KB)
        for (int i = 0; i < 32; ++i) {
            int id = i * 128 + t;                 // 0..4095
            int cc = id >> 3, kc = (id & 7) * 4;
            float4 f = *(const float4*)(Wo + (size_t)cc * 512 + k0 + kc);
            int ct = cc >> 4, n = cc & 15;
            {
                int k = kc;
                int lp = n + 16 * ((k >> 3) & 1);
                ldsB[ct * 256 + lp * 8 + ((k & 7) >> 1) + 4 * (k >> 4)] = packbf2(f.x, f.y);
            }
            {
                int k = kc + 2;
                int lp = n + 16 * ((k >> 3) & 1);
                ldsB[ct * 256 + lp * 8 + ((k & 7) >> 1) + 4 * (k >> 4)] = packbf2(f.z, f.w);
            }
        }
        __syncthreads();
        // A fragment straight from bf16 Ctx (row-major)
        const unsigned short* aptr = Ctx + (size_t)arow * 512 + k0;
        FragA A;
        A.q[0] = *(const u32x4*)(aptr + half * 8);
        A.q[1] = *(const u32x4*)(aptr + 16 + half * 8);
        #pragma unroll
        for (int c8 = 0; c8 < 8; ++c8) {
            const int gct = w * 8 + c8;
            const v16bf b = *(const v16bf*)((const unsigned short*)ldsB + gct * 512 + lane * 16);
            acc[c8] = wmma_bf16(A.v, b, acc[c8]);
        }
        __syncthreads();
    }

    // ---- epilogue: bias + residual into LDS row buffer ----
    #pragma unroll
    for (int c8 = 0; c8 < 8; ++c8) {
        const int c = w * 128 + c8 * 16 + n16;
        const float bb = bo[c];
        #pragma unroll
        for (int j = 0; j < 8; ++j) {
            int m = j + half * 8;
            float v = acc[c8][j] + bb + x[(size_t)(rbase + m) * 512 + c];
            ldsY[m * 512 + c] = v;
        }
    }
    __syncthreads();

    // ---- LayerNorm: 8 threads per row, 64 cols each ----
    const int rowm = t & 15;
    const int seg  = t >> 4;
    {
        float s = 0.f, s2 = 0.f;
        #pragma unroll 4
        for (int c = seg * 64; c < seg * 64 + 64; ++c) {
            float v = ldsY[rowm * 512 + c];
            s += v; s2 += v * v;
        }
        redS[rowm * 8 + seg]  = s;
        redS2[rowm * 8 + seg] = s2;
    }
    __syncthreads();
    if (t < 16) {
        float s = 0.f, s2 = 0.f;
        #pragma unroll
        for (int g = 0; g < 8; ++g) { s += redS[t * 8 + g]; s2 += redS2[t * 8 + g]; }
        float mu  = s * (1.0f / 512.0f);
        float var = s2 * (1.0f / 512.0f) - mu * mu;
        stats[t]      = mu;
        stats[16 + t] = rsqrtf(var + 1e-5f);
    }
    __syncthreads();
    {
        const float mu = stats[rowm];
        const float rs = stats[16 + rowm];
        const size_t gr = (size_t)(rbase + rowm) * 512;
        #pragma unroll 4
        for (int c = seg * 64; c < seg * 64 + 64; ++c) {
            float v = ldsY[rowm * 512 + c];
            out[gr + c] = (v - mu) * rs * gamma[c] + beta[c];
        }
    }
}

// ---------------------------------------------------------------------------
extern "C" void kernel_launch(void* const* d_in, const int* in_sizes, int n_in,
                              void* d_out, int out_size, void* d_ws, size_t ws_size,
                              hipStream_t stream)
{
    const float* x     = (const float*)d_in[0];
    const float* Wq    = (const float*)d_in[1];
    const float* bq    = (const float*)d_in[2];
    const float* Wk    = (const float*)d_in[3];
    const float* bk    = (const float*)d_in[4];
    const float* Wv    = (const float*)d_in[5];
    const float* bv    = (const float*)d_in[6];
    const float* Wo    = (const float*)d_in[7];
    const float* bo    = (const float*)d_in[8];
    const float* gamma = (const float*)d_in[9];
    const float* beta  = (const float*)d_in[10];

    const size_t NEL = (size_t)65536 * 512;            // elements per tensor
    unsigned short* ws  = (unsigned short*)d_ws;
    unsigned short* Qb  = ws;
    unsigned short* Kb  = ws + NEL;
    unsigned short* Vt  = ws + 2 * NEL;
    unsigned short* Ctx = ws + 3 * NEL;

    dim3 g1(1024, 8, 3);      // 65536/64 row tiles, 512/64 col tiles, Q/K/V
    qkv_proj_kernel<<<g1, 128, 0, stream>>>(x, Wq, bq, Wk, bk, Wv, bv, Qb, Kb, Vt);

    dim3 g2(1024, 8);         // (bt*8+h), query tile of 64
    attn_kernel<<<g2, 128, 0, stream>>>(Qb, Kb, Vt, Ctx);

    oproj_ln_kernel<<<4096, 128, 0, stream>>>(Ctx, Wo, bo, x, gamma, beta, (float*)d_out);
}
